// LSTM_76141180223684
// MI455X (gfx1250) — compile-verified, matrix operand order checked by reading, not run on hardware
//
#include <hip/hip_runtime.h>
#include <hip/hip_bf16.h>

// ---------------------------------------------------------------------------
// LSTM on MI455X (gfx1250): bf16 WMMA (f32 acc) for both the big input
// projection GEMM and the sequential recurrence.
//   I=256, H=512, B=32, T=2048.
// Gate-column layout (2048 cols): col(g,u) = (u>>5)*128 + g*32 + (u&31)
//   -> scan workgroup w owns units [32w,32w+32) = cols [128w,128w+128):
//      all 4 gates of its units are WG-local for the elementwise update, and
//      its hy output is exactly K-step ks==w of the next step's A fragment,
//      so h is exchanged directly in bf16 WMMA-fragment layout.
// ---------------------------------------------------------------------------

typedef __attribute__((ext_vector_type(16))) __bf16 v16bf;
typedef __attribute__((ext_vector_type(8)))  float  v8f;
typedef unsigned short u16b;

#define T_SEQ 2048
#define NWG   16      // workgroups in the scan kernel (grid-sync'd)

// ---- ws layout (bytes) ----
#define OFF_XG    ((size_t)0)                       // fp32 [65536][2048] = 512 MB
#define OFF_XA    (OFF_XG  + (size_t)536870912)     // bf16 A-frags of x: [4096 mt][8 ks][512]
#define OFF_WIB   (OFF_XA  + (size_t)33554432)      // bf16 B-frags of Wi: [128 nt][8 ks][512]
#define OFF_WHB   (OFF_WIB + (size_t)1048576)       // bf16 B-frags of Wh: [16 w][8 cg][16 ks][512]
#define OFF_BSUM  (OFF_WHB + (size_t)2097152)       // fp32 [2048] combined biases
#define OFF_HA    (OFF_BSUM+ (size_t)8192)          // bf16 ping-pong h A-frags [2][2 rb][16 ks][512]
#define OFF_CTR   (OFF_HA  + (size_t)65536)         // u32 grid-sync counter

// float -> bf16 bits, round-to-nearest-even
__device__ __forceinline__ u16b f2bf(float f) {
  union { float f; unsigned u; } x; x.f = f;
  unsigned u = x.u;
  return (u16b)((u + 0x7FFFu + ((u >> 16) & 1u)) >> 16);
}

__device__ __forceinline__ float sigmoidf_(float x) {
  return 1.0f / (1.0f + __expf(-x));
}

// Fragment layout (16-bit A/B operands, ISA 7.12.2), per 32-wide K-step:
//   lane l (m or n = l&15, hi = l>>4) holds 16 elements e:
//     e in [0,8)  -> k = hi*8 + e
//     e in [8,16) -> k = 16 + hi*8 + (e-8)
// i.e. two contiguous 8-element K runs per lane.

// ---------------------------------------------------------------------------
// K0a: biases, initial h (packed to A-fragment layout), grid-sync counter
// ---------------------------------------------------------------------------
__global__ void init_misc(const float* __restrict__ bii, const float* __restrict__ bif,
                          const float* __restrict__ bic, const float* __restrict__ bio,
                          const float* __restrict__ bhi, const float* __restrict__ bhf,
                          const float* __restrict__ bhc, const float* __restrict__ bho,
                          const float* __restrict__ hx,
                          float* __restrict__ bsum, u16b* __restrict__ hA,
                          unsigned* __restrict__ ctr) {
  int i = blockIdx.x * 256 + threadIdx.x;
  if (i < 2048) {
    int w = i >> 7, rem = i & 127, g = rem >> 5, u = (w << 5) | (rem & 31);
    const float* bi = (g == 0) ? bii : (g == 1) ? bif : (g == 2) ? bic : bio;
    const float* bh = (g == 0) ? bhi : (g == 1) ? bhf : (g == 2) ? bhc : bho;
    bsum[i] = bi[u] + bh[u];
  }
  int j = i - 2048;
  if (j >= 0 && j < 32 * 512) {          // pack hx -> hA buffer 0
    int b = j >> 9, k = j & 511;
    int ks = k >> 5, kk = k & 31;
    int l = (b & 15) | (((kk >> 3) & 1) << 4);
    int e = ((kk & 16) >> 1) + (kk & 7);
    hA[(b >> 4) * 8192 + ks * 512 + l * 16 + e] = f2bf(hx[j]);
  }
  if (i == 0) *ctr = 0u;
}

// ---------------------------------------------------------------------------
// K0b: pack x into WMMA A-fragments (bf16). row r = t*32+b (time-major).
// One thread = one (mt,ks,lane): two contiguous 8-float runs -> 16 bf16.
// ---------------------------------------------------------------------------
__global__ __launch_bounds__(256)
void pack_x(const float* __restrict__ x, u16b* __restrict__ xA) {
  int idx = blockIdx.x * 256 + threadIdx.x;   // < 4096*8*32
  int l  = idx & 31;
  int ks = (idx >> 5) & 7;
  int mt = idx >> 8;
  int m = l & 15, hi = l >> 4;
  int r = mt * 16 + m;
  int t = r >> 5, b = r & 31;
  const float* src = x + ((size_t)b * 2048 + t) * 256 + ks * 32 + hi * 8;
  u16b* dst = xA + ((size_t)mt * 8 + ks) * 512 + l * 16;
#pragma unroll
  for (int j = 0; j < 8; ++j) dst[j]     = f2bf(src[j]);       // e=j    k=hi*8+j
#pragma unroll
  for (int j = 0; j < 8; ++j) dst[8 + j] = f2bf(src[16 + j]);  // e=8+j  k=16+hi*8+j
}

// ---------------------------------------------------------------------------
// K0c: pack input weights (4x [512,256]) into B-fragments (gate-column layout)
// ---------------------------------------------------------------------------
__global__ __launch_bounds__(256)
void pack_wi(const float* __restrict__ Wii, const float* __restrict__ Wif,
             const float* __restrict__ Wic, const float* __restrict__ Wio,
             u16b* __restrict__ WiB) {
  int idx = blockIdx.x * 256 + threadIdx.x;   // < 128*8*32
  int l = idx & 31, ks = (idx >> 5) & 7, nt = idx >> 8;
  int n = l & 15, hi = l >> 4;
  int col = nt * 16 + n;
  int w = col >> 7, rem = col & 127, g = rem >> 5, u = (w << 5) | (rem & 31);
  const float* W = (g == 0) ? Wii : (g == 1) ? Wif : (g == 2) ? Wic : Wio;
  const float* src = W + (size_t)u * 256 + ks * 32 + hi * 8;
  u16b* dst = WiB + ((size_t)nt * 8 + ks) * 512 + l * 16;
#pragma unroll
  for (int j = 0; j < 8; ++j) dst[j]     = f2bf(src[j]);
#pragma unroll
  for (int j = 0; j < 8; ++j) dst[8 + j] = f2bf(src[16 + j]);
}

// ---------------------------------------------------------------------------
// K0d: pack hidden weights (4x [512,512]) into B-fragments, per-WG slices
// ---------------------------------------------------------------------------
__global__ __launch_bounds__(256)
void pack_wh(const float* __restrict__ Whi, const float* __restrict__ Whf,
             const float* __restrict__ Whc, const float* __restrict__ Who,
             u16b* __restrict__ WhB) {
  int idx = blockIdx.x * 256 + threadIdx.x;   // < 16*8*16*32
  int l  = idx & 31;
  int ks = (idx >> 5) & 15;
  int cg = (idx >> 9) & 7;
  int w  = idx >> 12;
  int n = l & 15, hi = l >> 4;
  int lc = cg * 16 + n;                 // local col in [0,128)
  int g = lc >> 5, ul = lc & 31, u = 32 * w + ul;
  const float* W = (g == 0) ? Whi : (g == 1) ? Whf : (g == 2) ? Whc : Who;
  const float* src = W + (size_t)u * 512 + ks * 32 + hi * 8;
  u16b* dst = WhB + (((size_t)(w * 8 + cg)) * 16 + ks) * 512 + l * 16;
#pragma unroll
  for (int j = 0; j < 8; ++j) dst[j]     = f2bf(src[j]);
#pragma unroll
  for (int j = 0; j < 8; ++j) dst[8 + j] = f2bf(src[16 + j]);
}

// ---------------------------------------------------------------------------
// K1: input projection GEMM  XG[65536,2048] = x·Wi4^T + (bi+bh)
// Register-blocked: one wave = 2(M) x 4(N) tiles of 16x16, K=256 = 8 k-steps,
// 8 WMMAs per A-pair. XG stores are non-temporal (write-once 512 MB stream).
// ---------------------------------------------------------------------------
__global__ __launch_bounds__(256)
void xgemm(const u16b* __restrict__ xA, const u16b* __restrict__ WiB,
           const float* __restrict__ bsum, float* __restrict__ XG) {
  int wv = threadIdx.x >> 5, lane = threadIdx.x & 31;
  int gw = blockIdx.x * 8 + wv;           // 65536 waves
  int ntb = gw & 31, mtb = gw >> 5;       // 4 N-tiles, 2 M-tiles per wave
  const u16b* aBase = xA  + (size_t)(mtb * 2) * (8 * 512) + lane * 16;
  const u16b* bBase = WiB + (size_t)(ntb * 4) * (8 * 512) + lane * 16;
  v8f acc[2][4] = {};
#pragma unroll 2
  for (int ks = 0; ks < 8; ++ks) {
    v16bf a0 = *(const v16bf*)(const void*)(aBase + (0 * 8 + ks) * 512);
    v16bf a1 = *(const v16bf*)(const void*)(aBase + (1 * 8 + ks) * 512);
#pragma unroll
    for (int j = 0; j < 4; ++j) {
      v16bf bv = *(const v16bf*)(const void*)(bBase + (j * 8 + ks) * 512);
      acc[0][j] = __builtin_amdgcn_wmma_f32_16x16x32_bf16(false, a0, false, bv,
                                                          (short)0, acc[0][j], false, false);
      acc[1][j] = __builtin_amdgcn_wmma_f32_16x16x32_bf16(false, a1, false, bv,
                                                          (short)0, acc[1][j], false, false);
    }
  }
  int N = lane & 15;
  int mBase = (lane < 16) ? 0 : 8;
#pragma unroll
  for (int mi = 0; mi < 2; ++mi)
#pragma unroll
    for (int j = 0; j < 4; ++j) {
      int col = (ntb * 4 + j) * 16 + N;
      float bias = bsum[col];
      float* dst = XG + (size_t)((mtb * 2 + mi) * 16) * 2048 + col;
#pragma unroll
      for (int i = 0; i < 8; ++i)
        __builtin_nontemporal_store(acc[mi][j][i] + bias,
                                    dst + (size_t)(mBase + i) * 2048);
    }
}

// ---------------------------------------------------------------------------
// K2: persistent recurrence. 16 WGs x 512 threads (16 waves each).
// - Wh fragment slice (128 KB) cached in LDS once; B = ds_load_b128.
// - h exchanged as bf16 A-fragments via global ping-pong (L2-hot 32 KB),
//   written directly by the elementwise phase (ks == WG id), no re-staging.
// - XG (HBM stream) software-pipelined one step ahead: its latency hides
//   under the elementwise phase + barrier spin instead of the serial path.
// - one device-scope arrive+spin barrier per step; cell state lives in LDS.
// ---------------------------------------------------------------------------
__global__ __launch_bounds__(512)
void lstm_scan(const float* __restrict__ XG, const u16b* __restrict__ WhB,
               const float* __restrict__ cx, u16b* __restrict__ hA,
               unsigned* __restrict__ ctr, float* __restrict__ out) {
  __shared__ __align__(32) u16b sWh[8 * 16 * 512];  // Wh B-frags (128 KB)
  __shared__ float sPre[32][129];                   // gate preacts, padded (16.1 KB)
  __shared__ float sC[32][32];                      // cell state [ul][b] (4 KB)

  const int w    = blockIdx.x;
  const int tid  = threadIdx.x;
  const int wv   = tid >> 5;
  const int lane = tid & 31;
  const int cg   = wv >> 1;     // col-group 0..7 (16 cols each)
  const int rb   = wv & 1;      // batch row block (0: b 0-15, 1: b 16-31)

  // preload this WG's Wh fragment slice into LDS (one-time 128 KB)
  {
    const uint4* src = (const uint4*)(const void*)(WhB + (size_t)w * (8 * 16 * 512));
    uint4* dst = (uint4*)(void*)sWh;
    for (int i = tid; i < (8 * 16 * 512) / 8; i += 512) dst[i] = src[i];
  }
  // init cell-state slice from cx (sC[ul][b] so elementwise lanes = b)
  for (int i = tid; i < 32 * 32; i += 512) {
    int ul = i >> 5, b = i & 31;
    sC[ul][b] = cx[b * 512 + 32 * w + ul];
  }
  __syncthreads();

  float* outBase = out;
  float* hnOut = out + (size_t)512 * 65536;
  float* cnOut = hnOut + 32 * 512;
  float* igOut = cnOut + 32 * 512;
  float* fgOut = igOut + 32 * 512;
  float* ogOut = fgOut + 32 * 512;

  // per-wave accumulator tile coordinates
  const int N     = lane & 15;
  const int mBase = (lane < 16) ? 0 : 8;
  const int bRow  = rb * 16 + mBase;            // first batch row of acc elems
  const float* xgBase = XG + (size_t)bRow * 2048 + (w * 128 + cg * 16 + N);

  // prefetch XG for t = 0
  float xg[8];
#pragma unroll
  for (int i = 0; i < 8; ++i)
    xg[i] = __builtin_nontemporal_load(xgBase + (size_t)i * 2048);

  for (int t = 0; t < T_SEQ; ++t) {
    const u16b* hR = hA + (size_t)(t & 1) * 16384;        // read frags
    u16b*       hW = hA + (size_t)((t + 1) & 1) * 16384;  // write frags

    // 16x16 tile: rows = batch block rb, cols = 16 gate-cols of group cg
    v8f acc = {};
#pragma unroll
    for (int ks = 0; ks < 16; ++ks) {
      v16bf av = *(const v16bf*)(const void*)(hR + rb * 8192 + ks * 512 + lane * 16);
      v16bf bv = *(const v16bf*)(const void*)(&sWh[(cg * 16 + ks) * 512 + lane * 16]);
      acc = __builtin_amdgcn_wmma_f32_16x16x32_bf16(false, av, false, bv,
                                                    (short)0, acc, false, false);
    }

    // combine with prefetched x-projection (+both biases), stash to LDS
#pragma unroll
    for (int i = 0; i < 8; ++i)
      sPre[bRow + i][cg * 16 + N] = acc[i] + xg[i];

    // prefetch XG for t+1 (issued last in LOADcnt order: only next step's
    // combine waits on it; latency hidden by elementwise + barrier spin)
    if (t + 1 < T_SEQ) {
      const float* p = xgBase + (size_t)(t + 1) * 65536;
#pragma unroll
      for (int i = 0; i < 8; ++i)
        xg[i] = __builtin_nontemporal_load(p + (size_t)i * 2048);
    }
    __syncthreads();

    // elementwise LSTM cell update (all 4 gates WG-local); lanes = b so the
    // 128 MB of `out` stores are fully coalesced. hy is also written directly
    // as bf16 A-fragment elements of the next step's K-step ks == w.
    for (int i = tid; i < 32 * 32; i += 512) {
      int ul = i >> 5, b = i & 31;
      float ig  = sigmoidf_(sPre[b][ul]);
      float fg  = sigmoidf_(sPre[b][32 + ul]);
      float cg_ = tanhf(sPre[b][64 + ul]);
      float og  = sigmoidf_(sPre[b][96 + ul]);
      float cy  = fg * sC[ul][b] + ig * cg_;
      float hy  = og * tanhf(cy);
      sC[ul][b] = cy;
      int l = (b & 15) | (((ul >> 3) & 1) << 4);
      int e = ((ul & 16) >> 1) + (ul & 7);
      hW[(b >> 4) * 8192 + w * 512 + l * 16 + e] = f2bf(hy);
      int u = 32 * w + ul;
      __builtin_nontemporal_store(hy, outBase + (size_t)u * 65536 + t * 32 + b);
      if (t == T_SEQ - 1) {
        hnOut[b * 512 + u] = hy; cnOut[b * 512 + u] = cy;
        igOut[b * 512 + u] = ig; fgOut[b * 512 + u] = fg; ogOut[b * 512 + u] = og;
      }
    }

    // device-scope arrive + spin barrier across the 16 WGs
    __threadfence();
    __syncthreads();
    if (tid == 0) {
      atomicAdd(ctr, 1u);
      unsigned target = (unsigned)(t + 1) * NWG;
      while (__hip_atomic_load(ctr, __ATOMIC_RELAXED, __HIP_MEMORY_SCOPE_AGENT) < target)
        __builtin_amdgcn_s_sleep(1);
    }
    __syncthreads();
    __threadfence();
  }
}

// ---------------------------------------------------------------------------
extern "C" void kernel_launch(void* const* d_in, const int* in_sizes, int n_in,
                              void* d_out, int out_size, void* d_ws, size_t ws_size,
                              hipStream_t stream) {
  const float* x   = (const float*)d_in[0];
  const float* hx  = (const float*)d_in[1];
  const float* cx  = (const float*)d_in[2];
  const float* Wii = (const float*)d_in[3];  const float* bii = (const float*)d_in[4];
  const float* Wif = (const float*)d_in[5];  const float* bif = (const float*)d_in[6];
  const float* Wic = (const float*)d_in[7];  const float* bic = (const float*)d_in[8];
  const float* Wio = (const float*)d_in[9];  const float* bio = (const float*)d_in[10];
  const float* Whi = (const float*)d_in[11]; const float* bhi = (const float*)d_in[12];
  const float* Whf = (const float*)d_in[13]; const float* bhf = (const float*)d_in[14];
  const float* Whc = (const float*)d_in[15]; const float* bhc = (const float*)d_in[16];
  const float* Who = (const float*)d_in[17]; const float* bho = (const float*)d_in[18];

  char* ws = (char*)d_ws;
  float*    XG   = (float*)(ws + OFF_XG);
  u16b*     xA   = (u16b*)(ws + OFF_XA);
  u16b*     WiB  = (u16b*)(ws + OFF_WIB);
  u16b*     WhB  = (u16b*)(ws + OFF_WHB);
  float*    bsum = (float*)(ws + OFF_BSUM);
  u16b*     hA   = (u16b*)(ws + OFF_HA);
  unsigned* ctr  = (unsigned*)(ws + OFF_CTR);
  float*    out  = (float*)d_out;

  init_misc<<<72, 256, 0, stream>>>(bii, bif, bic, bio, bhi, bhf, bhc, bho,
                                    hx, bsum, hA, ctr);
  pack_x <<<4096, 256, 0, stream>>>(x, xA);
  pack_wi<<<128,  256, 0, stream>>>(Wii, Wif, Wic, Wio, WiB);
  pack_wh<<<256,  256, 0, stream>>>(Whi, Whf, Whc, Who, WhB);
  xgemm  <<<8192, 256, 0, stream>>>(xA, WiB, bsum, XG);
  lstm_scan<<<NWG, 512, 0, stream>>>(XG, WhB, cx, hA, ctr, out);
}
